// onn_conv2d_35665408426209
// MI455X (gfx1250) — compile-verified, matrix operand order checked by reading, not run on hardware
//
#include <hip/hip_runtime.h>
#include <hip/hip_bf16.h>

typedef __attribute__((ext_vector_type(16))) _Float16 v16h;
typedef __attribute__((ext_vector_type(8)))  _Float16 v8h;
typedef __attribute__((ext_vector_type(8)))  float    v8f;

#define CIN   128
#define HH    32
#define WW    32
#define COUT  256
#define DTOT  1152       // CIN*9
#define RCH   18         // DTOT/64
#define LSP   1024       // HH*WW
#define NT    128        // spatial rows per block
#define MT    64         // channels per block
#define XSTR  72         // LDS row stride in f16 (64 + 8 pad -> 144B, 16B aligned, bank-friendly)

// ---- monotonic float<->uint keys so atomicMax/atomicMin on uint order floats ----
__device__ __forceinline__ unsigned f2key(float f) {
    unsigned b = __float_as_uint(f);
    return (b & 0x80000000u) ? ~b : (b | 0x80000000u);
}
__device__ __forceinline__ float key2f(unsigned k) {
    unsigned b = (k & 0x80000000u) ? (k & 0x7FFFFFFFu) : ~k;
    return __uint_as_float(b);
}

__global__ void onn_init_mm(unsigned* mm) {
    mm[0] = 0u;            // running max key
    mm[1] = 0xFFFFFFFFu;   // running min key
}

__global__ __launch_bounds__(256)
void onn_minmax(const float* __restrict__ w, unsigned* __restrict__ mm, int n) {
    __shared__ unsigned smax[256], smin[256];
    const int tid = threadIdx.x;
    unsigned vmax = 0u, vmin = 0xFFFFFFFFu;
    for (int i = blockIdx.x * 256 + tid; i < n; i += gridDim.x * 256) {
        unsigned k = f2key(w[i]);
        vmax = vmax > k ? vmax : k;
        vmin = vmin < k ? vmin : k;
    }
    smax[tid] = vmax; smin[tid] = vmin;
    __syncthreads();
    for (int s = 128; s > 0; s >>= 1) {
        if (tid < s) {
            smax[tid] = smax[tid] > smax[tid + s] ? smax[tid] : smax[tid + s];
            smin[tid] = smin[tid] < smin[tid + s] ? smin[tid] : smin[tid + s];
        }
        __syncthreads();
    }
    if (tid == 0) {
        atomicMax(&mm[0], smax[0]);
        atomicMin(&mm[1], smin[0]);
    }
}

__global__ __launch_bounds__(256)
void onn_quant(const float* __restrict__ w, const unsigned* __restrict__ mm,
               _Float16* __restrict__ wq, int n) {
    float wmax = key2f(mm[0]);
    float wmin = key2f(mm[1]);
    float scale = 15.0f / (wmax - wmin + 1e-9f);   // (2^BITS - 1) / range
    int i = blockIdx.x * 256 + threadIdx.x;
    if (i < n) {
        float q = rintf(w[i] * scale);             // round-half-even == jnp.round
        q = fminf(fmaxf(q, -7.0f), 7.0f);          // clamp to +/- QMAX
        wq[i] = (_Float16)q;
    }
}

__global__ __launch_bounds__(256)
void onn_main(const float* __restrict__ inp, const _Float16* __restrict__ wq,
              float* __restrict__ out) {
    __shared__ __align__(16) _Float16 xs[NT * XSTR];   // 128 x 64 activations (im2col)
    __shared__ __align__(16) _Float16 wsh[MT * XSTR];  //  64 x 64 quantized weights

    const int tid  = threadIdx.x;
    const int mt   = blockIdx.x & 3;        // 4 channel tiles
    const int nt   = blockIdx.x >> 2;       // 128 spatial tiles
    const int m0   = mt * MT;
    const int n0   = nt * NT;
    const int b    = n0 >> 10;              // batch (NT divides 1024)
    const int sp0  = n0 & 1023;             // spatial offset inside image

    const int wave  = tid >> 5;
    const int lane  = tid & 31;
    const int mrow  = (wave & 3) * 16;      // channel subtile base within block
    const int nbase = (wave >> 2) * 64;     // spatial base within block (0 or 64)
    const int lrow  = lane & 15;
    const int koff  = (lane < 16) ? 0 : 8;  // ISA 16-bit A/B lane K pattern
    const int nsel  = lane >> 4;

    v8f cnt[4];
    #pragma unroll
    for (int j = 0; j < 4; ++j) cnt[j] = (v8f){};

    for (int r = 0; r < RCH; ++r) {
        // ---- stage weight tile: 64 rows x 64 K, b128 vectorized ----
        {
            int t = tid;
            #pragma unroll
            for (int it = 0; it < 2; ++it) {
                int ml = t >> 3;            // 0..63
                int kc = (t & 7) * 8;       // 0..56
                const v8h* gp = (const v8h*)(wq + (size_t)(m0 + ml) * DTOT + r * 64 + kc);
                *(v8h*)(&wsh[ml * XSTR + kc]) = *gp;
                t += 256;
            }
        }
        // ---- stage activation tile with on-the-fly im2col: 128 rows x 64 K ----
        #pragma unroll 4
        for (int e = tid; e < NT * 64; e += 256) {
            int nl  = e & (NT - 1);         // consecutive lanes -> consecutive spatial (coalesced)
            int kk  = e >> 7;
            int sp  = sp0 + nl;
            int h   = sp >> 5;
            int w   = sp & 31;
            int d   = r * 64 + kk;
            int cin = d / 9;
            int rem = d - cin * 9;
            int kh  = rem / 3;
            int kw  = rem - kh * 3;
            int ih  = h + kh - 1;
            int iw  = w + kw - 1;
            float v = 0.0f;
            if ((unsigned)ih < 32u && (unsigned)iw < 32u)
                v = inp[(((size_t)b * CIN + cin) * HH + ih) * WW + iw];
            xs[nl * XSTR + kk] = (_Float16)v;
        }
        __syncthreads();

        // ---- per-chunk GEMM: fresh f32 accumulators, 2 WMMAs (K=64) per subtile ----
        v8f acc[4];
        #pragma unroll
        for (int j = 0; j < 4; ++j) acc[j] = (v8f){};

        #pragma unroll
        for (int half = 0; half < 2; ++half) {
            const int kb = half * 32;
            // A fragment: weights, row = channel
            v16h a;
            {
                const _Float16* ap = &wsh[(mrow + lrow) * XSTR + kb + koff];
                v8h lo = *(const v8h*)(ap);        // K = kb+koff .. +7
                v8h hi = *(const v8h*)(ap + 16);   // K = kb+koff+16 .. +23
                #pragma unroll
                for (int i = 0; i < 8; ++i) { a[i] = lo[i]; a[8 + i] = hi[i]; }
            }
            // B fragments: activations, col = spatial
            #pragma unroll
            for (int j = 0; j < 4; ++j) {
                const _Float16* bp = &xs[(nbase + j * 16 + lrow) * XSTR + kb + koff];
                v8h lo = *(const v8h*)(bp);
                v8h hi = *(const v8h*)(bp + 16);
                v16h bb;
                #pragma unroll
                for (int i = 0; i < 8; ++i) { bb[i] = lo[i]; bb[8 + i] = hi[i]; }
                acc[j] = __builtin_amdgcn_wmma_f32_16x16x32_f16(
                    false, a, false, bb, (short)0, acc[j], false, false);
            }
        }

        // ---- binarize (s<0 -> 0 else 1, NaN -> 1 as in jnp.where) + count ----
        #pragma unroll
        for (int j = 0; j < 4; ++j)
            #pragma unroll
            for (int i = 0; i < 8; ++i)
                cnt[j][i] += (acc[j][i] < 0.0f) ? 0.0f : 1.0f;

        __syncthreads();
    }

    // ---- store: out[b, channel, spatial]; lane dim = spatial -> coalesced ----
    #pragma unroll
    for (int j = 0; j < 4; ++j) {
        int spc = sp0 + nbase + j * 16 + lrow;
        #pragma unroll
        for (int i = 0; i < 8; ++i) {
            int mchan = m0 + mrow + i + 8 * nsel;   // D layout: VGPR i, lanes>=16 -> M+8
            out[((size_t)b * COUT + mchan) * LSP + spc] = cnt[j][i];
        }
    }
}

extern "C" void kernel_launch(void* const* d_in, const int* in_sizes, int n_in,
                              void* d_out, int out_size, void* d_ws, size_t ws_size,
                              hipStream_t stream) {
    const float* inp = (const float*)d_in[0];   // [16,128,32,32]
    const float* wt  = (const float*)d_in[1];   // [256,128,3,3]
    float* out = (float*)d_out;                 // [16,256,32,32]

    unsigned*  mm = (unsigned*)d_ws;                          // [2] min/max keys
    _Float16*  wq = (_Float16*)((char*)d_ws + 256);           // [256*1152] f16 quantized

    const int nw = in_sizes[1];                 // 294912

    onn_init_mm<<<1, 1, 0, stream>>>(mm);
    onn_minmax<<<256, 256, 0, stream>>>(wt, mm, nw);
    onn_quant<<<(nw + 255) / 256, 256, 0, stream>>>(wt, mm, wq, nw);

    // 4 channel tiles x 128 spatial tiles = 512 blocks, 256 threads (8 wave32)
    onn_main<<<512, 256, 0, stream>>>(inp, wq, out);
}
